// SingleStreamBlockLoraProcessor_50440095924239
// MI455X (gfx1250) — matrix operand order repaired and checked by reading, not compiled, not run on hardware
//
#include <hip/hip_runtime.h>
#include <hip/hip_bf16.h>

// ---------------------------------------------------------------------------
// CDNA5 (gfx1250) single-stream transformer block with LoRA.
// All large matmuls (lin1, lin2, Q.K^T, P.V) use v_wmma_f32_16x16x32_bf16.
// Weights converted once to bf16 + transposed (N-major): bf16 copies
// (132MB + 94MB) fit the 192MB L2 across the GEMMs.
// GEMM waves compute 32x64 tiles; K-loop is a ping-pong unroll-by-2 so each
// fragment set is written by loads and consumed by WMMAs directly (no
// register-copy swap), with __launch_bounds__ sized so nothing spills.
// Attention: per-wave flash kernel computing S^T (query on lane axis ->
// lane-local softmax stats), V loads hoisted above softmax VALU work.
// ---------------------------------------------------------------------------

#define DIM      3072
#define HEADS    24
#define HEAD_DIM 128
#define MLPD     12288
#define LSEQ     2048
#define N1       (3*DIM + MLPD)   // 21504
#define N2       DIM              // 3072
#define K2       (DIM + MLPD)     // 15360
#define RANK     16

typedef __attribute__((ext_vector_type(16))) __bf16 v16bf;
typedef __attribute__((ext_vector_type(8)))  float  v8f;

__device__ __forceinline__ unsigned short f2bf(float f) {
    unsigned int u = __float_as_uint(f);
    u += 0x7fffu + ((u >> 16) & 1u);           // round-to-nearest-even
    return (unsigned short)(u >> 16);
}
__device__ __forceinline__ unsigned int pack_bf(float lo, float hi) {
    return (unsigned int)f2bf(lo) | ((unsigned int)f2bf(hi) << 16);
}
// Build a 16-element bf16 fragment from two 16-byte chunks.
__device__ __forceinline__ v16bf load_frag2(const unsigned short* p0,
                                            const unsigned short* p1) {
    union { v16bf v; uint4 q[2]; } u;
    u.q[0] = *reinterpret_cast<const uint4*>(p0);
    u.q[1] = *reinterpret_cast<const uint4*>(p1);
    return u.v;
}
__device__ __forceinline__ v8f wmma_bf16(v16bf a, v16bf b, v8f c) {
    return __builtin_amdgcn_wmma_f32_16x16x32_bf16(false, a, false, b,
                                                   (short)0, c, false, false);
}
__device__ __forceinline__ float gelu_tanh(float x) {
    return 0.5f * x * (1.0f + tanhf(0.7978845608028654f * (x + 0.044715f * x * x * x)));
}

// ---------------------------------------------------------------------------
// Ping-pong WMMA core: one wave computes a 32x64 output tile.
// acc[t*4+s] = (m-subtile t, n-subtile s). 8 WMMAs per 32-deep K chunk,
// two K chunks in flight (sets A/B) -> no register-copy buffer swap.
// ---------------------------------------------------------------------------
struct FragSet { v16bf a0, a1, b0, b1, b2, b3; };

struct TilePtrs {
    const unsigned short* arow0;
    const unsigned short* arow1;
    const unsigned short* bcol0;
    const unsigned short* bcol1;
    const unsigned short* bcol2;
    const unsigned short* bcol3;
    int khalf;
};

__device__ __forceinline__ void load_set(const TilePtrs& p, int k0, FragSet& f) {
    f.a0 = load_frag2(p.arow0 + k0 + p.khalf, p.arow0 + k0 + 8 + p.khalf);
    f.a1 = load_frag2(p.arow1 + k0 + p.khalf, p.arow1 + k0 + 8 + p.khalf);
    f.b0 = load_frag2(p.bcol0 + k0, p.bcol0 + k0 + 8);
    f.b1 = load_frag2(p.bcol1 + k0, p.bcol1 + k0 + 8);
    f.b2 = load_frag2(p.bcol2 + k0, p.bcol2 + k0 + 8);
    f.b3 = load_frag2(p.bcol3 + k0, p.bcol3 + k0 + 8);
}
__device__ __forceinline__ void fma_set(const FragSet& f, v8f acc[8]) {
    acc[0] = wmma_bf16(f.a0, f.b0, acc[0]);
    acc[1] = wmma_bf16(f.a0, f.b1, acc[1]);
    acc[2] = wmma_bf16(f.a0, f.b2, acc[2]);
    acc[3] = wmma_bf16(f.a0, f.b3, acc[3]);
    acc[4] = wmma_bf16(f.a1, f.b0, acc[4]);
    acc[5] = wmma_bf16(f.a1, f.b1, acc[5]);
    acc[6] = wmma_bf16(f.a1, f.b2, acc[6]);
    acc[7] = wmma_bf16(f.a1, f.b3, acc[7]);
}

template <int K>
__device__ __forceinline__ void wmma_core_32x64(const unsigned short* __restrict__ A,
                                                const unsigned short* __restrict__ BT,
                                                int mBase, int nBase, int lane,
                                                v8f acc[8]) {
    TilePtrs p;
    p.khalf = (lane >> 4) * 16;
    const int ksel = (lane >> 4) * 16;
    p.arow0 = A + (size_t)(mBase + (lane & 15)) * K;
    p.arow1 = p.arow0 + (size_t)16 * K;
    p.bcol0 = BT + (size_t)(nBase +  0 + (lane & 15)) * K + ksel;
    p.bcol1 = BT + (size_t)(nBase + 16 + (lane & 15)) * K + ksel;
    p.bcol2 = BT + (size_t)(nBase + 32 + (lane & 15)) * K + ksel;
    p.bcol3 = BT + (size_t)(nBase + 48 + (lane & 15)) * K + ksel;

    FragSet fa, fb;
    load_set(p, 0, fa);
    #pragma unroll 1
    for (int k0 = 0; k0 + 64 < K; k0 += 64) {
        load_set(p, k0 + 32, fb);   // overlaps the 8 WMMAs on set A
        fma_set(fa, acc);
        load_set(p, k0 + 64, fa);   // overlaps the 8 WMMAs on set B
        fma_set(fb, acc);
    }
    load_set(p, K - 32, fb);
    fma_set(fa, acc);
    fma_set(fb, acc);
}

// ---------------------------------------------------------------------------
// K0: fp32 -> bf16 convert + transpose (w[k,n] -> wT[n,k]), 32x32 LDS tiles.
// ---------------------------------------------------------------------------
__global__ void convt_bf16_kernel(const float* __restrict__ w,
                                  unsigned short* __restrict__ wT,
                                  int K, int N) {
    __shared__ unsigned short tile[32][33];
    const int n0 = blockIdx.x * 32, k0 = blockIdx.y * 32;
    const int tx = threadIdx.x & 31, ty = threadIdx.x >> 5;   // 256 threads
    #pragma unroll
    for (int i = ty; i < 32; i += 8)
        tile[i][tx] = f2bf(w[(size_t)(k0 + i) * N + n0 + tx]);
    __syncthreads();
    #pragma unroll
    for (int i = ty; i < 32; i += 8)
        wT[(size_t)(n0 + i) * K + k0 + tx] = tile[tx][i];
}

// ---------------------------------------------------------------------------
// K1: mod = silu(vec) @ mod_w + mod_b   (1x3072 @ 3072x9216)
// ---------------------------------------------------------------------------
__global__ void mod_gemv_kernel(const float* __restrict__ vec,
                                const float* __restrict__ mw,
                                const float* __restrict__ mb,
                                float* __restrict__ mod) {
    __shared__ float sv[DIM];
    for (int i = threadIdx.x; i < DIM; i += 256) {
        float v = vec[i];
        sv[i] = v / (1.0f + __expf(-v));
    }
    __syncthreads();
    const int j = blockIdx.x * 256 + threadIdx.x;
    float acc = mb[j];
    for (int i = 0; i < DIM; ++i) acc += sv[i] * mw[(size_t)i * (3*DIM) + j];
    mod[j] = acc;
}

// ---------------------------------------------------------------------------
// K2: x_mod = (1+scale)*layernorm(x) + shift, stored fp32 + bf16.
// ---------------------------------------------------------------------------
__global__ void ln_mod_kernel(const float* __restrict__ x,
                              const float* __restrict__ mod,
                              float* __restrict__ xm,
                              unsigned short* __restrict__ xmb) {
    __shared__ float red[256];
    const int l = blockIdx.x;
    const float* row = x + (size_t)l * DIM;
    float s = 0.f;
    for (int c = threadIdx.x; c < DIM; c += 256) s += row[c];
    red[threadIdx.x] = s; __syncthreads();
    for (int o = 128; o > 0; o >>= 1) {
        if (threadIdx.x < o) red[threadIdx.x] += red[threadIdx.x + o];
        __syncthreads();
    }
    const float mu = red[0] * (1.0f / DIM);
    __syncthreads();
    float vs = 0.f;
    for (int c = threadIdx.x; c < DIM; c += 256) { float d = row[c] - mu; vs += d * d; }
    red[threadIdx.x] = vs; __syncthreads();
    for (int o = 128; o > 0; o >>= 1) {
        if (threadIdx.x < o) red[threadIdx.x] += red[threadIdx.x + o];
        __syncthreads();
    }
    const float rstd = rsqrtf(red[0] * (1.0f / DIM) + 1e-6f);
    for (int c = threadIdx.x; c < DIM; c += 256) {
        float v = (1.0f + mod[DIM + c]) * ((row[c] - mu) * rstd) + mod[c];
        xm[(size_t)l * DIM + c] = v;
        xmb[(size_t)l * DIM + c] = f2bf(v);
    }
}

// ---------------------------------------------------------------------------
// K3: GEMM1  h = x_mod(2048x3072) @ w1T^T + b1.
// Epilogue splits qkv (fp32) vs gelu(mlp) (bf16 -> lin2in).
// ---------------------------------------------------------------------------
__global__ void __launch_bounds__(128, 2)
gemm1_kernel(const unsigned short* __restrict__ A,   // 2048xK
             const unsigned short* __restrict__ BT,  // N-major NxK
             const float* __restrict__ bias,
             float* __restrict__ qkv,                // 2048x9216
             unsigned short* __restrict__ lin2in) {  // 2048x15360
    const int lane  = threadIdx.x & 31;
    const int wave  = threadIdx.x >> 5;
    const int mBase = blockIdx.y * 32;
    const int nBase = blockIdx.x * 256 + wave * 64;

    v8f acc[8];
    #pragma unroll
    for (int i = 0; i < 8; ++i)
        #pragma unroll
        for (int e = 0; e < 8; ++e) acc[i][e] = 0.f;

    wmma_core_32x64<DIM>(A, BT, mBase, nBase, lane, acc);

    #pragma unroll
    for (int t = 0; t < 2; ++t)
        #pragma unroll
        for (int s = 0; s < 4; ++s) {
            const int col = nBase + s * 16 + (lane & 15);
            #pragma unroll
            for (int r = 0; r < 8; ++r) {
                const int row = mBase + t * 16 + r + 8 * (lane >> 4);
                float v = acc[t * 4 + s][r] + bias[col];
                if (col < 3 * DIM) {
                    qkv[(size_t)row * (3 * DIM) + col] = v;
                } else {
                    lin2in[(size_t)row * K2 + DIM + (col - 3 * DIM)] = f2bf(gelu_tanh(v));
                }
            }
        }
}

// ---------------------------------------------------------------------------
// K7: GEMM2  out = lin2in(2048x15360) @ w2T^T + b2  (fp32 out)
// ---------------------------------------------------------------------------
__global__ void __launch_bounds__(128, 2)
gemm2_kernel(const unsigned short* __restrict__ A,   // 2048x15360
             const unsigned short* __restrict__ BT,  // 3072x15360
             const float* __restrict__ bias,
             float* __restrict__ C) {                // 2048x3072
    const int lane  = threadIdx.x & 31;
    const int wave  = threadIdx.x >> 5;
    const int mBase = blockIdx.y * 32;
    const int nBase = blockIdx.x * 256 + wave * 64;

    v8f acc[8];
    #pragma unroll
    for (int i = 0; i < 8; ++i)
        #pragma unroll
        for (int e = 0; e < 8; ++e) acc[i][e] = 0.f;

    wmma_core_32x64<K2>(A, BT, mBase, nBase, lane, acc);

    #pragma unroll
    for (int t = 0; t < 2; ++t)
        #pragma unroll
        for (int s = 0; s < 4; ++s) {
            const int col = nBase + s * 16 + (lane & 15);
            #pragma unroll
            for (int r = 0; r < 8; ++r) {
                const int row = mBase + t * 16 + r + 8 * (lane >> 4);
                C[(size_t)row * DIM + col] = acc[t * 4 + s][r] + bias[col];
            }
        }
}

// ---------------------------------------------------------------------------
// K4a/K8a: rank-16 LoRA down: tmp(2048x16) = X(2048x3072) @ down(3072x16)
// ---------------------------------------------------------------------------
__global__ void lora_down_kernel(const float* __restrict__ X,
                                 const float* __restrict__ down,
                                 float* __restrict__ tmp) {
    __shared__ float red[256];
    const int l = blockIdx.x;
    const int r = threadIdx.x & 15, slice = threadIdx.x >> 4;
    float acc = 0.f;
    for (int k = slice; k < DIM; k += 16)
        acc += X[(size_t)l * DIM + k] * down[k * RANK + r];
    red[threadIdx.x] = acc; __syncthreads();
    if (threadIdx.x < 16) {
        float s = 0.f;
        #pragma unroll
        for (int i = 0; i < 16; ++i) s += red[threadIdx.x + 16 * i];
        tmp[l * RANK + threadIdx.x] = s;
    }
}

// K4b: dst += tmp @ up   (up: 16 x N)
__global__ void lora_up_add_kernel(const float* __restrict__ tmp,
                                   const float* __restrict__ up,
                                   float* __restrict__ dst, int N) {
    const int l = blockIdx.y;
    const int n = blockIdx.x * 256 + threadIdx.x;
    float s = 0.f;
    #pragma unroll
    for (int r = 0; r < RANK; ++r) s += tmp[l * RANK + r] * up[(size_t)r * N + n];
    dst[(size_t)l * N + n] += s;
}

// ---------------------------------------------------------------------------
// K5: per-(l,h) rmsnorm(q,k) + rope; emit q,k as (H,L,D) bf16, v as (H,D,L).
// ---------------------------------------------------------------------------
__global__ void qkv_norm_rope_kernel(const float* __restrict__ qkv,
                                     const float* __restrict__ qs,
                                     const float* __restrict__ ks,
                                     const float* __restrict__ pe,
                                     unsigned short* __restrict__ qb,
                                     unsigned short* __restrict__ kb,
                                     unsigned short* __restrict__ vT) {
    const int l = blockIdx.x, h = blockIdx.y, d = threadIdx.x;  // 128 threads
    const float* base = qkv + (size_t)l * (3 * DIM) + h * HEAD_DIM;
    float q = base[d], k = base[DIM + d], v = base[2 * DIM + d];

    __shared__ float sq[HEAD_DIM], sk[HEAD_DIM], wsum[2][4];
    float q2 = q * q, k2 = k * k;
    #pragma unroll
    for (int o = 16; o >= 1; o >>= 1) {
        q2 += __shfl_xor(q2, o, 32);
        k2 += __shfl_xor(k2, o, 32);
    }
    if ((threadIdx.x & 31) == 0) { wsum[0][d >> 5] = q2; wsum[1][d >> 5] = k2; }
    __syncthreads();
    const float qss = wsum[0][0] + wsum[0][1] + wsum[0][2] + wsum[0][3];
    const float kss = wsum[1][0] + wsum[1][1] + wsum[1][2] + wsum[1][3];
    const float qr = rsqrtf(qss * (1.0f / HEAD_DIM) + 1e-6f);
    const float kr = rsqrtf(kss * (1.0f / HEAD_DIM) + 1e-6f);
    sq[d] = q * qr * qs[d];
    sk[d] = k * kr * ks[d];
    __syncthreads();
    const int i = d >> 1, a = d & 1;
    const float* pp = pe + ((size_t)(l * (HEAD_DIM / 2) + i) * 2 + a) * 2;
    const float qo = pp[0] * sq[2 * i] + pp[1] * sq[2 * i + 1];
    const float ko = pp[0] * sk[2 * i] + pp[1] * sk[2 * i + 1];
    qb[((size_t)h * LSEQ + l) * HEAD_DIM + d] = f2bf(qo);
    kb[((size_t)h * LSEQ + l) * HEAD_DIM + d] = f2bf(ko);
    vT[((size_t)h * HEAD_DIM + d) * LSEQ + l] = f2bf(v);
}

// ---------------------------------------------------------------------------
// K6: flash attention, one wave per (head, 16-query tile).
// S^T = K.Q^T via WMMA -> query on lane axis -> lane-local softmax stats.
// K-frags batched into one clause; V-frag loads hoisted above softmax VALU.
// ---------------------------------------------------------------------------
__global__ void __launch_bounds__(128, 1)
attn_kernel(const unsigned short* __restrict__ qb,
            const unsigned short* __restrict__ kbuf,
            const unsigned short* __restrict__ vT,
            unsigned short* __restrict__ lin2in) {
    const int lane = threadIdx.x & 31;
    const int wid  = (blockIdx.x * blockDim.x + threadIdx.x) >> 5;
    const int h    = wid >> 7;       // 128 query tiles per head
    const int qt   = wid & 127;
    const unsigned short* qh = qb   + (size_t)h * LSEQ * HEAD_DIM;
    const unsigned short* kh = kbuf + (size_t)h * LSEQ * HEAD_DIM;
    const unsigned short* vh = vT   + (size_t)h * HEAD_DIM * LSEQ;

    const int qi     = lane & 15;
    const int dsel   = (lane >> 4) * 8;
    const int khalf8 = (lane >> 4) * 8;

    v16bf qf[4];
    #pragma unroll
    for (int c = 0; c < 4; ++c) {
        const unsigned short* p = qh + (size_t)(qt * 16 + qi) * HEAD_DIM + c * 32 + dsel;
        qf[c] = load_frag2(p, p + 16);
    }

    v8f acc[8];
    #pragma unroll
    for (int od = 0; od < 8; ++od)
        #pragma unroll
        for (int e = 0; e < 8; ++e) acc[od][e] = 0.f;

    float m = -3.0e38f, lsum = 0.f;
    const float scale = 0.08838834764831845f;   // 1/sqrt(128)
    const bool lowhalf = lane < 16;

    #pragma unroll 1
    for (int kb0 = 0; kb0 < LSEQ; kb0 += 32) {
        // Batch all 8 K-fragments (2 key tiles x 4 d-chunks) into one clause.
        v16bf kf[8];
        #pragma unroll
        for (int c = 0; c < 4; ++c) {
            const unsigned short* k0p =
                kh + (size_t)(kb0 + (lane & 15)) * HEAD_DIM + c * 32 + khalf8;
            kf[c] = load_frag2(k0p, k0p + 16);
            const unsigned short* k1p = k0p + 16 * HEAD_DIM;
            kf[4 + c] = load_frag2(k1p, k1p + 16);
        }
        v8f st0, st1;
        #pragma unroll
        for (int e = 0; e < 8; ++e) { st0[e] = 0.f; st1[e] = 0.f; }
        #pragma unroll
        for (int c = 0; c < 4; ++c) st0 = wmma_bf16(kf[c], qf[c], st0);
        #pragma unroll
        for (int c = 0; c < 4; ++c) st1 = wmma_bf16(kf[4 + c], qf[c], st1);

        // Issue V-fragment loads now; latency overlaps the softmax VALU below.
        v16bf vf[8];
        #pragma unroll
        for (int od = 0; od < 8; ++od) {
            const unsigned short* vp =
                vh + (size_t)(od * 16 + (lane & 15)) * LSEQ + kb0 + (lane >> 4) * 16;
            vf[od] = load_frag2(vp, vp + 8);
        }

        // Online softmax; query i = lane&15, stats lane-local + half-swap.
        float p0[8], p1[8];
        float lm = -3.0e38f;
        #pragma unroll
        for (int r = 0; r < 8; ++r) {
            p0[r] = st0[r] * scale;
            p1[r] = st1[r] * scale;
            lm = fmaxf(lm, fmaxf(p0[r], p1[r]));
        }
        const float rowm = fmaxf(lm, __shfl_xor(lm, 16, 32));
        const float newm = fmaxf(m, rowm);
        const float corr = __expf(m - newm);
        float ps = 0.f;
        #pragma unroll
        for (int r = 0; r < 8; ++r) {
            p0[r] = __expf(p0[r] - newm);
            p1[r] = __expf(p1[r] - newm);
            ps += p0[r] + p1[r];
        }
        ps += __shfl_xor(ps, 16, 32);
        lsum = lsum * corr + ps;
        m = newm;
        #pragma unroll
        for (int od = 0; od < 8; ++od)
            #pragma unroll
            for (int e = 0; e < 8; ++e) acc[od][e] *= corr;

        // Repack P^T (C layout) into B-operand layout: swap key rows 8..15
        // and 16..23 across lane halves.
        float o0[8], o1[8];
        #pragma unroll
        for (int r = 0; r < 8; ++r) {
            o0[r] = __shfl_xor(p0[r], 16, 32);
            o1[r] = __shfl_xor(p1[r], 16, 32);
        }
        union { v16bf v; unsigned int w[8]; } pb;
        #pragma unroll
        for (int vv = 0; vv < 4; ++vv)
            pb.w[vv] = lowhalf ? pack_bf(p0[2 * vv], p0[2 * vv + 1])
                               : pack_bf(o1[2 * vv], o1[2 * vv + 1]);
        #pragma unroll
        for (int vv = 4; vv < 8; ++vv) {
            const int s = 2 * (vv - 4);
            pb.w[vv] = lowhalf ? pack_bf(o0[s], o0[s + 1])
                               : pack_bf(p1[s], p1[s + 1]);
        }
        // O^T += V^T . P^T
        #pragma unroll
        for (int od = 0; od < 8; ++od)
            acc[od] = wmma_bf16(vf[od], pb.v, acc[od]);
    }

    const float inv = 1.0f / lsum;
    const int qrow = qt * 16 + (lane & 15);
    #pragma unroll
    for (int od = 0; od < 8; ++od)
        #pragma unroll
        for (int r = 0; r < 8; ++r) {
            const int dd = od * 16 + r + 8 * (lane >> 4);
            lin2in[(size_t)qrow * K2 + h * HEAD_DIM + dd] = f2bf(acc[od][r] * inv);
        }
}

// ---------------------------------------------------------------------------
// K9: y = x + gate * (out + tmp2 @ proj_up)
// ---------------------------------------------------------------------------
__global__ void final_kernel(const float* __restrict__ x,
                             const float* __restrict__ mod,
                             const float* __restrict__ outb,
                             const float* __restrict__ tmp2,
                             const float* __restrict__ pup,
                             float* __restrict__ y) {
    const int l = blockIdx.y;
    const int c = blockIdx.x * 256 + threadIdx.x;
    float s = 0.f;
    #pragma unroll
    for (int r = 0; r < RANK; ++r) s += tmp2[l * RANK + r] * pup[(size_t)r * DIM + c];
    const float o = outb[(size_t)l * DIM + c] + s;
    y[(size_t)l * DIM + c] = x[(size_t)l * DIM + c] + mod[2 * DIM + c] * o;
}

// ---------------------------------------------------------------------------
extern "C" void kernel_launch(void* const* d_in, const int* in_sizes, int n_in,
                              void* d_out, int out_size, void* d_ws, size_t ws_size,
                              hipStream_t stream) {
    (void)in_sizes; (void)n_in; (void)out_size; (void)ws_size;
    const float* x      = (const float*)d_in[0];
    const float* vec    = (const float*)d_in[1];
    const float* pe     = (const float*)d_in[2];
    const float* mod_w  = (const float*)d_in[3];
    const float* mod_b  = (const float*)d_in[4];
    const float* lin1_w = (const float*)d_in[5];
    const float* lin1_b = (const float*)d_in[6];
    const float* q_sc   = (const float*)d_in[7];
    const float* k_sc   = (const float*)d_in[8];
    const float* lin2_w = (const float*)d_in[9];
    const float* lin2_b = (const float*)d_in[10];
    const float* qdn    = (const float*)d_in[11];
    const float* qup    = (const float*)d_in[12];
    const float* pdn    = (const float*)d_in[13];
    const float* pup    = (const float*)d_in[14];
    float* out = (float*)d_out;

    char* wp = (char*)d_ws;
    auto alloc = [&](size_t bytes) -> char* {
        char* p = wp;
        wp += (bytes + 255) & ~(size_t)255;
        return p;
    };
    float*          modv   = (float*)alloc((size_t)3 * DIM * 4);
    float*          xm     = (float*)alloc((size_t)LSEQ * DIM * 4);
    unsigned short* xmb    = (unsigned short*)alloc((size_t)LSEQ * DIM * 2);
    unsigned short* w1T    = (unsigned short*)alloc((size_t)N1 * DIM * 2);
    unsigned short* w2T    = (unsigned short*)alloc((size_t)N2 * K2 * 2);
    float*          qkv    = (float*)alloc((size_t)LSEQ * 3 * DIM * 4);
    float*          tmp    = (float*)alloc((size_t)LSEQ * RANK * 4);
    unsigned short* qbuf   = (unsigned short*)alloc((size_t)HEADS * LSEQ * HEAD_DIM * 2);
    unsigned short* kbuf   = (unsigned short*)alloc((size_t)HEADS * LSEQ * HEAD_DIM * 2);
    unsigned short* vTb    = (unsigned short*)alloc((size_t)HEADS * HEAD_DIM * LSEQ * 2);
    unsigned short* lin2in = (unsigned short*)alloc((size_t)LSEQ * K2 * 2);
    float*          outb   = (float*)alloc((size_t)LSEQ * DIM * 4);
    float*          tmp2   = (float*)alloc((size_t)LSEQ * RANK * 4);

    // Weight convert+transpose to bf16 (once per call; bf16 copies fit L2).
    convt_bf16_kernel<<<dim3(N1 / 32, DIM / 32), 256, 0, stream>>>(lin1_w, w1T, DIM, N1);
    convt_bf16_kernel<<<dim3(N2 / 32, K2 / 32), 256, 0, stream>>>(lin2_w, w2T, K2, N2);

    // Modulation + layernorm.
    mod_gemv_kernel<<<(3 * DIM) / 256, 256, 0, stream>>>(vec, mod_w, mod_b, modv);
    ln_mod_kernel<<<LSEQ, 256, 0, stream>>>(x, modv, xm, xmb);

    // lin1 GEMM (WMMA, 32x64 per wave) -> qkv fp32 + gelu(mlp) bf16.
    gemm1_kernel<<<dim3(N1 / 256, LSEQ / 32), 128, 0, stream>>>(xmb, w1T, lin1_b, qkv, lin2in);

    // qkv LoRA.
    lora_down_kernel<<<LSEQ, 256, 0, stream>>>(xm, qdn, tmp);
    lora_up_add_kernel<<<dim3((3 * DIM) / 256, LSEQ), 256, 0, stream>>>(tmp, qup, qkv, 3 * DIM);

    // RMSNorm + RoPE, layout for attention.
    qkv_norm_rope_kernel<<<dim3(LSEQ, HEADS), HEAD_DIM, 0, stream>>>(
        qkv, q_sc, k_sc, pe, qbuf, kbuf, vTb);

    // Flash attention (WMMA): 24 heads * 128 query tiles = 3072 waves.
    attn_kernel<<<(HEADS * (LSEQ / 16) * 32) / 128, 128, 0, stream>>>(qbuf, kbuf, vTb, lin2in);

    // lin2 GEMM (WMMA).
    gemm2_kernel<<<dim3(N2 / 256, LSEQ / 32), 128, 0, stream>>>(lin2in, w2T, lin2_b, outb);

    // proj LoRA + gated residual.
    lora_down_kernel<<<LSEQ, 256, 0, stream>>>(outb, pdn, tmp2);
    final_kernel<<<dim3(DIM / 256, LSEQ), 256, 0, stream>>>(x, modv, outb, tmp2, pup, out);
}
